// Decoder_76802605187101
// MI455X (gfx1250) — matrix operand order repaired and checked by reading, not compiled, not run on hardware
//
#include <hip/hip_runtime.h>
#include <math.h>

#define Bdim 256
#define Ndim 400
#define Hdim 512
#define Edim 256
#define Vdim 50000
#define OOVdim 50
#define Tdim 49

typedef __attribute__((ext_vector_type(16))) __bf16 v16bf;
typedef __attribute__((ext_vector_type(2)))  __bf16 v2bf;
typedef __attribute__((ext_vector_type(8)))  float  v8f;

union FragU { v16bf v; unsigned short u[16]; unsigned p32[8]; };

// Convert a pair of floats to a packed bf16 dword. Prefer the packed-cvt
// builtin; otherwise rely on the backend lowering of fptrunc f32->bf16
// (selects v_cvt_pk_bf16_f32 on gfx1250-class hardware).
__device__ __forceinline__ unsigned pk2(float x, float y) {
#if __has_builtin(__builtin_amdgcn_cvt_pk_bf16_f32)
    union { v2bf v; unsigned u; } r;
    r.v = __builtin_amdgcn_cvt_pk_bf16_f32(x, y);
    return r.u;
#else
    union { v2bf v; unsigned u; } r;
    r.v[0] = (__bf16)x;
    r.v[1] = (__bf16)y;
    return r.u;
#endif
}

__device__ __forceinline__ void pack4(FragU& f, int o, float4 q) {   // o = dword offset
    f.p32[o]     = pk2(q.x, q.y);
    f.p32[o + 1] = pk2(q.z, q.w);
}

// A fragment: 16x32 bf16 tile of row-major fp32 A[R,K], rows rowTile..+15, k0..k0+31.
// lane<16 -> M=lane, K=k0..k0+15; lanes>=16 -> M=lane-16, K=k0+16..k0+31.
__device__ __forceinline__ v16bf load_frag_a(const float* __restrict__ A, int rowTile,
                                             int K, int k0, int maxRow) {
    int lane = threadIdx.x & 31;
    int m = rowTile + (lane & 15);
    if (m > maxRow) m = maxRow;
    const float4* p4 = (const float4*)(A + (long long)m * K + k0 + ((lane & 16) ? 16 : 0));
    float4 q0 = p4[0], q1 = p4[1], q2 = p4[2], q3 = p4[3];   // p[0..15]
    FragU f;
    pack4(f, 0, q0); pack4(f, 2, q1); pack4(f, 4, q2); pack4(f, 6, q3);
    return f.v;
}

// B fragment: 32x16 bf16 tile from row-major W[Nrows,K] (B = W^T).
// lane&15 = column n; lanes<16 hold K=k0..k0+15, lanes>=16 hold k0+16..k0+31.
__device__ __forceinline__ v16bf load_frag_b(const float* __restrict__ W, int colTile,
                                             int K, int k0, int maxCol) {
    int lane = threadIdx.x & 31;
    int n = colTile + (lane & 15);
    if (n > maxCol) n = maxCol;
    const float4* p4 = (const float4*)(W + (long long)n * K + k0 + ((lane & 16) ? 16 : 0));
    float4 q0 = p4[0], q1 = p4[1], q2 = p4[2], q3 = p4[3];
    FragU f;
    pack4(f, 0, q0); pack4(f, 2, q1); pack4(f, 4, q2); pack4(f, 6, q3);
    return f.v;
}

#define WMMA_BF16(a, b, c) \
    __builtin_amdgcn_wmma_f32_16x16x32_bf16(false, (a), false, (b), (short)0, (c), false, false)

// ---------------------------------------------------------------------------
// Generic NT GEMM: C[M,N](ldc) = (ACC ? C : 0) + bias + A[M,K] @ W[N,K]^T
// Block = 8 waves: 2 row-groups x 4 col-groups. Each wave owns 32x64 output
// (2 row-tiles x 4 col-tiles = 8 accumulators): per K-step 2 A-frags + 4
// B-frags -> 8 wmma. A reused 4x, B reused 2x.
// ---------------------------------------------------------------------------
template <int ACC>
__global__ __launch_bounds__(256)
void gemm_nt_wmma(const float* __restrict__ A, const float* __restrict__ W,
                  const float* __restrict__ bias, float* __restrict__ C,
                  int M, int N, int K, int ldc)
{
    int wave = threadIdx.x >> 5;
    int lane = threadIdx.x & 31;
    int rowBase = blockIdx.x * 64 + (wave >> 2) * 32;   // row tiles: +0, +16
    int colBase = blockIdx.y * 256 + (wave & 3) * 64;   // col tiles: +0,16,32,48

    v8f acc[2][4] = {};
    for (int k0 = 0; k0 < K; k0 += 32) {
        if (k0 + 32 < K)
            __builtin_prefetch(A + (long long)rowBase * K + k0 + 32, 0, 0);
        v16bf a0 = load_frag_a(A, rowBase,      K, k0, M - 1);
        v16bf a1 = load_frag_a(A, rowBase + 16, K, k0, M - 1);
#pragma unroll
        for (int t = 0; t < 4; ++t) {
            v16bf b = load_frag_b(W, colBase + t * 16, K, k0, N - 1);
            acc[0][t] = WMMA_BF16(a0, b, acc[0][t]);
            acc[1][t] = WMMA_BF16(a1, b, acc[1][t]);
        }
    }

    int colLane = lane & 15;
#pragma unroll
    for (int rt = 0; rt < 2; ++rt) {
        int rbase = rowBase + rt * 16 + ((lane & 16) ? 8 : 0);
        bool rowsOK = (rbase + 7) < M;
#pragma unroll
        for (int t = 0; t < 4; ++t) {
            int col = colBase + t * 16 + colLane;
            if (col >= N) continue;
            float bv = (bias != nullptr) ? bias[col] : 0.0f;
            float* cp = C + (long long)rbase * ldc + col;
            if (rowsOK) {
#pragma unroll
                for (int i = 0; i < 8; ++i) {
                    float v = acc[rt][t][i] + bv;
                    if (ACC) v += cp[(long long)i * ldc];
                    cp[(long long)i * ldc] = v;
                }
            } else {
                for (int i = 0; i < 8 && (rbase + i) < M; ++i) {
                    float v = acc[rt][t][i] + bv;
                    if (ACC) v += cp[(long long)i * ldc];
                    cp[(long long)i * ldc] = v;
                }
            }
        }
    }
}

// ---------------------------------------------------------------------------
// Fused attention score: scores[r] = sum_e v[e]*tanh( (A @ W^T)[r,e] + spart[b(r),e] )
// Block covers 64 rows; 8 waves = 2 row-groups (32 rows) x 4 column groups.
// Each wave: per 64-col chunk, 2 A-frags + 4 B-frags -> 8 wmma, then tanh/v
// reduction into lane partials; shuffle + LDS float atomics fold columns.
// ---------------------------------------------------------------------------
__global__ __launch_bounds__(256)
void attn_score_wmma(const float* __restrict__ A, const float* __restrict__ W,
                     const float* __restrict__ spart, const float* __restrict__ vvec,
                     float* __restrict__ scores,
                     int R, int K, int Dout, int rows_per_b)
{
    __shared__ float srow[64];
    int wave = threadIdx.x >> 5;
    int lane = threadIdx.x & 31;
    if (threadIdx.x < 64) srow[threadIdx.x] = 0.0f;
    __syncthreads();

    int rowBase = blockIdx.x * 64 + (wave >> 2) * 32;   // two 16-row tiles
    int cg = wave & 3;                                  // column group 0..3
    int nChunks = Dout >> 6;                            // 64 columns per chunk
    int colLane = lane & 15;

    // per-element row -> batch mapping (independent of column chunk)
    int bb[2][8];
#pragma unroll
    for (int rt = 0; rt < 2; ++rt) {
        int rbase = rowBase + rt * 16 + ((lane & 16) ? 8 : 0);
#pragma unroll
        for (int i = 0; i < 8; ++i) {
            int r = rbase + i;
            if (r > R - 1) r = R - 1;
            bb[rt][i] = r / rows_per_b;
        }
    }

    v8f partial[2] = {};
    for (int ch = cg; ch < nChunks; ch += 4) {
        int colBase = ch * 64;
        v8f acc[2][4] = {};
        for (int k0 = 0; k0 < K; k0 += 32) {
            v16bf a0 = load_frag_a(A, rowBase,      K, k0, R - 1);
            v16bf a1 = load_frag_a(A, rowBase + 16, K, k0, R - 1);
#pragma unroll
            for (int t = 0; t < 4; ++t) {
                v16bf b = load_frag_b(W, colBase + t * 16, K, k0, Dout - 1);
                acc[0][t] = WMMA_BF16(a0, b, acc[0][t]);
                acc[1][t] = WMMA_BF16(a1, b, acc[1][t]);
            }
        }
#pragma unroll
        for (int t = 0; t < 4; ++t) {
            int col = colBase + t * 16 + colLane;
            float vw = vvec[col];
#pragma unroll
            for (int rt = 0; rt < 2; ++rt)
#pragma unroll
                for (int i = 0; i < 8; ++i)
                    partial[rt][i] +=
                        tanhf(acc[rt][t][i] + spart[(long long)bb[rt][i] * Dout + col]) * vw;
        }
    }
    // reduce the 16 column-lanes of each half, then fold into LDS
#pragma unroll
    for (int rt = 0; rt < 2; ++rt) {
#pragma unroll
        for (int i = 0; i < 8; ++i) {
            float v = partial[rt][i];
            v += __shfl_xor(v, 1, 32);
            v += __shfl_xor(v, 2, 32);
            v += __shfl_xor(v, 4, 32);
            v += __shfl_xor(v, 8, 32);
            partial[rt][i] = v;
        }
    }
    if ((lane & 15) == 0) {
        int rloc = (wave >> 2) * 32 + ((lane & 16) ? 8 : 0);
#pragma unroll
        for (int rt = 0; rt < 2; ++rt)
#pragma unroll
            for (int i = 0; i < 8; ++i)
                atomicAdd(&srow[rloc + rt * 16 + i], partial[rt][i]);   // ds_add_f32
    }
    __syncthreads();
    if (threadIdx.x < 64) {
        int r = blockIdx.x * 64 + threadIdx.x;
        if (r < R) scores[r] = srow[threadIdx.x];
    }
}

// ---------------------------------------------------------------------------
// Small elementwise / reduction kernels
// ---------------------------------------------------------------------------
__global__ void concat_xin_kernel(const float* __restrict__ x_t, const float* __restrict__ ct_e,
                                  float* __restrict__ xin) {
    int idx = blockIdx.x * blockDim.x + threadIdx.x;
    int total = Bdim * (Edim + 2 * Hdim);
    if (idx >= total) return;
    int w = Edim + 2 * Hdim;
    int b = idx / w, j = idx % w;
    xin[idx] = (j < Edim) ? x_t[b * Edim + j] : ct_e[b * 2 * Hdim + (j - Edim)];
}

__global__ void lstm_pointwise_kernel(const float* __restrict__ gates,
                                      const float* __restrict__ dec_c,
                                      float* __restrict__ h_new, float* __restrict__ c_new,
                                      float* __restrict__ sthat) {
    int idx = blockIdx.x * blockDim.x + threadIdx.x;
    if (idx >= Bdim * Hdim) return;
    int b = idx / Hdim, j = idx % Hdim;
    const float* g = gates + (long long)b * 4 * Hdim;
    float gi = 1.0f / (1.0f + expf(-g[j]));
    float gf = 1.0f / (1.0f + expf(-g[Hdim + j]));
    float gg = tanhf(g[2 * Hdim + j]);
    float go = 1.0f / (1.0f + expf(-g[3 * Hdim + j]));
    float c = gf * dec_c[idx] + gi * gg;
    float h = go * tanhf(c);
    h_new[idx] = h;
    c_new[idx] = c;
    sthat[b * 2 * Hdim + j] = h;
    sthat[b * 2 * Hdim + Hdim + j] = c;
}

// prev_s_new = cat(prev_s, h_new[:,None,:]) -- float4 vectorized copy
__global__ void prevs_new_kernel(const float4* __restrict__ prev_s,
                                 const float4* __restrict__ h_new,
                                 float4* __restrict__ prevs_new) {
    const int HW = Hdim / 4;
    int idx = blockIdx.x * blockDim.x + threadIdx.x;
    int total = Bdim * (Tdim + 1) * HW;
    if (idx >= total) return;
    int b = idx / ((Tdim + 1) * HW);
    int rem = idx % ((Tdim + 1) * HW);
    int t = rem / HW, j = rem % HW;
    prevs_new[idx] = (t < Tdim) ? prev_s[((long long)b * Tdim + t) * HW + j]
                                : h_new[b * HW + j];
}

__global__ void enc_norm_kernel(const float* __restrict__ et, const float* __restrict__ sum_t,
                                const float* __restrict__ mask,
                                float* __restrict__ sum_new, float* __restrict__ at) {
    __shared__ float red[256];
    int b = blockIdx.x, tid = threadIdx.x;
    float local = 0.0f;
    for (int n = tid; n < Ndim; n += 256) {
        int o = b * Ndim + n;
        float e  = expf(et[o]);
        float st = sum_t[o];
        sum_new[o] = st + e;
        float a = (e / st) * mask[o];
        at[o] = a;
        local += a;
    }
    red[tid] = local; __syncthreads();
    for (int off = 128; off > 0; off >>= 1) {
        if (tid < off) red[tid] += red[tid + off];
        __syncthreads();
    }
    float s = red[0];
    for (int n = tid; n < Ndim; n += 256) at[b * Ndim + n] /= s;
}

// ct_e_new[b,:] = sum_n at[b,n] * enc_out[b,n,:]  (float4, 256 thr x 4 cols = 1024)
__global__ void cte_kernel(const float* __restrict__ at, const float* __restrict__ enc_out,
                           float* __restrict__ cte) {
    int b = blockIdx.x, tid = threadIdx.x;
    const float4* eo = (const float4*)(enc_out + (long long)b * Ndim * (2 * Hdim));
    float4 acc = make_float4(0.f, 0.f, 0.f, 0.f);
    for (int n = 0; n < Ndim; ++n) {
        float a = at[b * Ndim + n];
        float4 v = eo[n * (2 * Hdim / 4) + tid];
        acc.x += a * v.x; acc.y += a * v.y; acc.z += a * v.z; acc.w += a * v.w;
    }
    ((float4*)(cte + (long long)b * 2 * Hdim))[tid] = acc;
}

__global__ void dec_softmax_kernel(const float* __restrict__ etd, float* __restrict__ atd) {
    __shared__ float red[64];
    int b = blockIdx.x, t = threadIdx.x;       // 64 threads
    float v = (t < Tdim) ? etd[b * Tdim + t] : -1e30f;
    red[t] = v; __syncthreads();
    for (int off = 32; off > 0; off >>= 1) {
        if (t < off) red[t] = fmaxf(red[t], red[t + off]);
        __syncthreads();
    }
    float m = red[0]; __syncthreads();
    float e = (t < Tdim) ? expf(v - m) : 0.0f;
    red[t] = e; __syncthreads();
    for (int off = 32; off > 0; off >>= 1) {
        if (t < off) red[t] += red[t + off];
        __syncthreads();
    }
    if (t < Tdim) atd[b * Tdim + t] = e / red[0];
}

// ct_d[b,:] = sum_t atd[b,t] * prev_s[b,t,:]  (float2, 256 thr x 2 cols = 512)
__global__ void ctd_kernel(const float* __restrict__ atd, const float* __restrict__ prev_s,
                           float* __restrict__ ctd) {
    int b = blockIdx.x, tid = threadIdx.x;
    const float2* ps = (const float2*)(prev_s + (long long)b * Tdim * Hdim);
    float2 acc = make_float2(0.f, 0.f);
    for (int t = 0; t < Tdim; ++t) {
        float a = atd[b * Tdim + t];
        float2 v = ps[t * (Hdim / 2) + tid];
        acc.x += a * v.x; acc.y += a * v.y;
    }
    ((float2*)(ctd + (long long)b * Hdim))[tid] = acc;
}

__global__ void pgen_kernel(const float* __restrict__ cte, const float* __restrict__ ctd,
                            const float* __restrict__ h_new, const float* __restrict__ c_new,
                            const float* __restrict__ x, const float* __restrict__ pw,
                            const float* __restrict__ pb, float* __restrict__ p_gen) {
    __shared__ float red[256];
    int b = blockIdx.x, tid = threadIdx.x;
    const int W = 5 * Hdim + Edim;  // 2816: [ct_e(1024) | ct_d(512) | h(512) | c(512) | x(256)]
    float local = 0.0f;
    for (int j = tid; j < W; j += 256) {
        float v;
        if (j < 1024)       v = cte[b * 1024 + j];
        else if (j < 1536)  v = ctd[b * 512 + (j - 1024)];
        else if (j < 2048)  v = h_new[b * 512 + (j - 1536)];
        else if (j < 2560)  v = c_new[b * 512 + (j - 2048)];
        else                v = x[b * 256 + (j - 2560)];
        local += pw[j] * v;
    }
    red[tid] = local; __syncthreads();
    for (int off = 128; off > 0; off >>= 1) {
        if (tid < off) red[tid] += red[tid + off];
        __syncthreads();
    }
    if (tid == 0) p_gen[b] = 1.0f / (1.0f + expf(-(red[0] + pb[0])));
}

__global__ void concat_vin_kernel(const float* __restrict__ h_new, const float* __restrict__ cte,
                                  const float* __restrict__ ctd, float* __restrict__ vin) {
    int idx = blockIdx.x * blockDim.x + threadIdx.x;
    int total = Bdim * 4 * Hdim;
    if (idx >= total) return;
    int b = idx / (4 * Hdim), j = idx % (4 * Hdim);
    float v;
    if (j < 512)       v = h_new[b * 512 + j];
    else if (j < 1536) v = cte[b * 1024 + (j - 512)];
    else               v = ctd[b * 512 + (j - 1536)];
    vin[idx] = v;
}

// In-place softmax of logits in final_dist[:, :V] scaled by p_gen; zero OOV tail. float2.
__global__ void vocab_softmax_kernel(float* __restrict__ fd, const float* __restrict__ p_gen) {
    __shared__ float red[256];
    int b = blockIdx.x, tid = threadIdx.x;
    float* L = fd + (long long)b * (Vdim + OOVdim);
    float2* L2 = (float2*)L;
    const int V2 = Vdim / 2;
    float m = -1e30f;
    for (int j = tid; j < V2; j += 256) {
        float2 q = L2[j];
        m = fmaxf(m, fmaxf(q.x, q.y));
    }
    red[tid] = m; __syncthreads();
    for (int off = 128; off > 0; off >>= 1) {
        if (tid < off) red[tid] = fmaxf(red[tid], red[tid + off]);
        __syncthreads();
    }
    m = red[0]; __syncthreads();
    float s = 0.0f;
    for (int j = tid; j < V2; j += 256) {
        float2 q = L2[j];
        s += expf(q.x - m) + expf(q.y - m);
    }
    red[tid] = s; __syncthreads();
    for (int off = 128; off > 0; off >>= 1) {
        if (tid < off) red[tid] += red[tid + off];
        __syncthreads();
    }
    s = red[0];
    float scale = p_gen[b] / s;
    for (int j = tid; j < V2; j += 256) {
        float2 q = L2[j];
        q.x = scale * expf(q.x - m);
        q.y = scale * expf(q.y - m);
        L2[j] = q;
    }
    if (tid < OOVdim) L[Vdim + tid] = 0.0f;
}

__global__ void scatter_add_kernel(float* __restrict__ fd, const int* __restrict__ idx,
                                   const float* __restrict__ at, const float* __restrict__ p_gen) {
    int i = blockIdx.x * blockDim.x + threadIdx.x;
    if (i >= Bdim * Ndim) return;
    int b = i / Ndim;
    float w = (1.0f - p_gen[b]) * at[i];
    atomicAdd(&fd[(long long)b * (Vdim + OOVdim) + idx[i]], w);
}

// ---------------------------------------------------------------------------
extern "C" void kernel_launch(void* const* d_in, const int* in_sizes, int n_in,
                              void* d_out, int out_size, void* d_ws, size_t ws_size,
                              hipStream_t stream) {
    (void)in_sizes; (void)n_in; (void)out_size; (void)ws_size;
    const float* x_t      = (const float*)d_in[0];
    const float* dec_h    = (const float*)d_in[1];
    const float* dec_c    = (const float*)d_in[2];
    const float* enc_out  = (const float*)d_in[3];
    const float* mask     = (const float*)d_in[4];
    const float* ct_e     = (const float*)d_in[5];
    const int*   ext_voc  = (const int*)d_in[7];
    const float* sum_t    = (const float*)d_in[8];
    const float* prev_s   = (const float*)d_in[9];
    const float* x_in_w   = (const float*)d_in[10];
    const float* x_in_b   = (const float*)d_in[11];
    const float* w_ih     = (const float*)d_in[12];
    const float* w_hh     = (const float*)d_in[13];
    const float* b_ih     = (const float*)d_in[14];
    const float* b_hh     = (const float*)d_in[15];
    const float* pgen_w   = (const float*)d_in[16];
    const float* pgen_b   = (const float*)d_in[17];
    const float* V_w      = (const float*)d_in[18];
    const float* V_b      = (const float*)d_in[19];
    const float* V1_w     = (const float*)d_in[20];
    const float* V1_b     = (const float*)d_in[21];
    const float* Wh_w     = (const float*)d_in[22];
    const float* Ws_w     = (const float*)d_in[23];
    const float* Ws_b     = (const float*)d_in[24];
    const float* v_w      = (const float*)d_in[25];
    const float* Wprev_w  = (const float*)d_in[26];
    const float* Wsd_w    = (const float*)d_in[27];
    const float* Wsd_b    = (const float*)d_in[28];
    const float* vd_w     = (const float*)d_in[29];

    // Output layout (floats, reference return order)
    float* out_final = (float*)d_out;                                  // [B, V+OOV]
    float* out_h     = out_final + (size_t)Bdim * (Vdim + OOVdim);     // [B, H]
    float* out_c     = out_h + (size_t)Bdim * Hdim;                    // [B, H]
    float* out_cte   = out_c + (size_t)Bdim * Hdim;                    // [B, 2H]
    float* out_sum   = out_cte + (size_t)Bdim * 2 * Hdim;              // [B, N]
    float* out_prevs = out_sum + (size_t)Bdim * Ndim;                  // [B, T+1, H]

    // Workspace layout (floats)
    float* ws = (float*)d_ws;
    float* xin     = ws;                          // B*1280
    float* xbuf    = xin     + Bdim * 1280;       // B*256
    float* gates   = xbuf    + Bdim * 256;        // B*2048
    float* sthat   = gates   + Bdim * 2048;       // B*1024
    float* wspart  = sthat   + Bdim * 1024;       // B*1024
    float* wsdpart = wspart  + Bdim * 1024;       // B*512
    float* et      = wsdpart + Bdim * 512;        // B*N
    float* at      = et      + Bdim * Ndim;       // B*N
    float* etd     = at      + Bdim * Ndim;       // B*T
    float* atd     = etd     + Bdim * Tdim;       // B*T
    float* ctd     = atd     + Bdim * Tdim;       // B*512
    float* vin     = ctd     + Bdim * 512;        // B*2048
    float* outbuf  = vin     + Bdim * 2048;       // B*512
    float* p_gen   = outbuf  + Bdim * 512;        // B

    dim3 blk(256);
    auto g1 = [](int n) { return dim3((n + 255) / 256); };
    auto gg = [](int M, int N) { return dim3((M + 63) / 64, (N + 255) / 256); };

    // 1) x = Linear(cat(x_t, ct_e))
    concat_xin_kernel<<<g1(Bdim * 1280), blk, 0, stream>>>(x_t, ct_e, xin);
    gemm_nt_wmma<0><<<gg(Bdim, Edim), blk, 0, stream>>>(xin, x_in_w, x_in_b, xbuf,
                                                        Bdim, Edim, 1280, Edim);
    // 2) LSTM gates = x@w_ih^T + b_ih + dec_h@w_hh^T + b_hh
    gemm_nt_wmma<0><<<gg(Bdim, 4 * Hdim), blk, 0, stream>>>(xbuf, w_ih, b_ih, gates,
                                                            Bdim, 4 * Hdim, Edim, 4 * Hdim);
    gemm_nt_wmma<1><<<gg(Bdim, 4 * Hdim), blk, 0, stream>>>(dec_h, w_hh, b_hh, gates,
                                                            Bdim, 4 * Hdim, Hdim, 4 * Hdim);
    // 3) pointwise LSTM -> h_new, c_new, st_hat; prev_s_new
    lstm_pointwise_kernel<<<g1(Bdim * Hdim), blk, 0, stream>>>(gates, dec_c, out_h, out_c, sthat);
    prevs_new_kernel<<<g1(Bdim * (Tdim + 1) * Hdim / 4), blk, 0, stream>>>(
        (const float4*)prev_s, (const float4*)out_h, (float4*)out_prevs);

    // 4) encoder intra-temporal attention
    gemm_nt_wmma<0><<<gg(Bdim, 2 * Hdim), blk, 0, stream>>>(sthat, Ws_w, Ws_b, wspart,
                                                            Bdim, 2 * Hdim, 2 * Hdim, 2 * Hdim);
    attn_score_wmma<<<dim3((Bdim * Ndim + 63) / 64), blk, 0, stream>>>(
        enc_out, Wh_w, wspart, v_w, et, Bdim * Ndim, 2 * Hdim, 2 * Hdim, Ndim);
    enc_norm_kernel<<<dim3(Bdim), blk, 0, stream>>>(et, sum_t, mask, out_sum, at);
    cte_kernel<<<dim3(Bdim), blk, 0, stream>>>(at, enc_out, out_cte);

    // 5) intra-decoder attention
    gemm_nt_wmma<0><<<gg(Bdim, Hdim), blk, 0, stream>>>(out_h, Wsd_w, Wsd_b, wsdpart,
                                                        Bdim, Hdim, Hdim, Hdim);
    attn_score_wmma<<<dim3((Bdim * Tdim + 63) / 64), blk, 0, stream>>>(
        prev_s, Wprev_w, wsdpart, vd_w, etd, Bdim * Tdim, Hdim, Hdim, Tdim);
    dec_softmax_kernel<<<dim3(Bdim), dim3(64), 0, stream>>>(etd, atd);
    ctd_kernel<<<dim3(Bdim), blk, 0, stream>>>(atd, prev_s, ctd);

    // 6) pointer-generator head
    pgen_kernel<<<dim3(Bdim), blk, 0, stream>>>(out_cte, ctd, out_h, out_c, xbuf,
                                                pgen_w, pgen_b, p_gen);
    concat_vin_kernel<<<g1(Bdim * 4 * Hdim), blk, 0, stream>>>(out_h, out_cte, ctd, vin);
    gemm_nt_wmma<0><<<gg(Bdim, Hdim), blk, 0, stream>>>(vin, V_w, V_b, outbuf,
                                                        Bdim, Hdim, 4 * Hdim, Hdim);
    // logits written directly into final_dist rows (ldc = V+OOV), softmaxed in place
    gemm_nt_wmma<0><<<gg(Bdim, Vdim), blk, 0, stream>>>(outbuf, V1_w, V1_b, out_final,
                                                        Bdim, Vdim, Hdim, Vdim + OOVdim);
    vocab_softmax_kernel<<<dim3(Bdim), blk, 0, stream>>>(out_final, p_gen);
    scatter_add_kernel<<<g1(Bdim * Ndim), blk, 0, stream>>>(out_final, ext_voc, at, p_gen);
}
